// AttentionMechanism_67585605370569
// MI455X (gfx1250) — matrix operand order repaired and checked by reading, not compiled, not run on hardware
//
#include <hip/hip_runtime.h>

typedef __attribute__((ext_vector_type(16))) __bf16 v16bf;
typedef __attribute__((ext_vector_type(8)))  __bf16 v8bf;
typedef __attribute__((ext_vector_type(2)))  __bf16 v2bf;
typedef __attribute__((ext_vector_type(8)))  float  v8f;

constexpr int B_BATCH    = 32;
constexpr int C_CH       = 1024;
constexpr int N_SP       = 784;          // 28*28
constexpr int BM         = 128;          // block rows
constexpr int BN         = 64;           // block cols
constexpr int LDS_STRIDE = 40;           // 32 K elems + pad, 80B rows (16B aligned)

// ---------------------------------------------------------------------------
// Fragment loaders (wave32, v_wmma_f32_16x16x32_bf16 operand layouts)
// A (16x32): lanes 0-15 -> M=lane, K in {k0..k0+7, k0+16..k0+23}, k0 = 8*(lane>=16)
// B (32x16): lanes 0-15 -> N=lane, same K order as A: K in {k0..k0+7, k0+16..k0+23}
//            (LDS holds B K-contiguous per column)
// ---------------------------------------------------------------------------
__device__ inline v16bf load_fragA(const __bf16* s, int rowBase, int lane) {
  const int m  = rowBase + (lane & 15);
  const int k0 = (lane >> 4) << 3;
  const __bf16* p = s + m * LDS_STRIDE;
  v8bf a = *(const v8bf*)(p + k0);
  v8bf b = *(const v8bf*)(p + k0 + 16);
  v16bf f;
#pragma unroll
  for (int i = 0; i < 8; ++i) { f[i] = a[i]; f[i + 8] = b[i]; }
  return f;
}

__device__ inline v16bf load_fragB(const __bf16* s, int colBase, int lane) {
  const int n  = colBase + (lane & 15);
  const int ks = (lane >> 4) << 3;
  const __bf16* p = s + n * LDS_STRIDE + ks;
  v8bf a = *(const v8bf*)(p);
  v8bf b = *(const v8bf*)(p + 16);
  v16bf f;
#pragma unroll
  for (int i = 0; i < 8; ++i) { f[i] = a[i]; f[i + 8] = b[i]; }
  return f;
}

// Split a float4 into hi/lo bf16 and store 4 K-contiguous elems as 2x b32.
__device__ inline void store_split4(__bf16* shi, __bf16* slo, int off, float4 v) {
  __bf16 h0 = (__bf16)v.x; __bf16 l0 = (__bf16)(v.x - (float)h0);
  __bf16 h1 = (__bf16)v.y; __bf16 l1 = (__bf16)(v.y - (float)h1);
  __bf16 h2 = (__bf16)v.z; __bf16 l2 = (__bf16)(v.z - (float)h2);
  __bf16 h3 = (__bf16)v.w; __bf16 l3 = (__bf16)(v.w - (float)h3);
  *(v2bf*)(shi + off)     = (v2bf){h0, h1};
  *(v2bf*)(shi + off + 2) = (v2bf){h2, h3};
  *(v2bf*)(slo + off)     = (v2bf){l0, l1};
  *(v2bf*)(slo + off + 2) = (v2bf){l2, l3};
}

// ---------------------------------------------------------------------------
// Kernel 1: energy[b] = Q Q^T   (Q = x[b] viewed as [C, N])
// grid (C/BN, C/BM, B), 128 threads = 4 waves; each wave: 32x64 = 2x4 frags.
// bf16x3 emulated-f32: hi*hi + hi*lo + lo*hi per K tile.
// ---------------------------------------------------------------------------
__global__ __launch_bounds__(128) void qqt_kernel(const float* __restrict__ x,
                                                  float* __restrict__ energy) {
  __shared__ __bf16 sAhi[BM * LDS_STRIDE];
  __shared__ __bf16 sAlo[BM * LDS_STRIDE];
  __shared__ __bf16 sBhi[BN * LDS_STRIDE];
  __shared__ __bf16 sBlo[BN * LDS_STRIDE];

  const int b = blockIdx.z;
  const float* Q = x + (size_t)b * C_CH * N_SP;
  float* E = energy + (size_t)b * C_CH * C_CH;
  const int rowBlk = blockIdx.y * BM;
  const int colBlk = blockIdx.x * BN;
  const int tid  = threadIdx.x;
  const int lane = tid & 31;
  const int wr   = (tid >> 5) * 32;      // wave row offset (4 waves x 32 rows)

  v8f acc[2][4] = {};

  const int KT = (N_SP + 31) / 32;       // 25; last tile has 16 valid k (4 float4)
  for (int kt = 0; kt < KT; ++kt) {
    const int kg = kt * 32;
    if (kt < KT - 1) {                   // uniform branch: unguarded fast path
#pragma unroll
      for (int i = 0; i < 8; ++i) {      // A tile: 128 rows x 8 float4
        const int idx = tid + i * 128;
        const int r = idx >> 3, q = idx & 7;
        float4 v = *(const float4*)(Q + (size_t)(rowBlk + r) * N_SP + kg + q * 4);
        store_split4(sAhi, sAlo, r * LDS_STRIDE + q * 4, v);
      }
#pragma unroll
      for (int i = 0; i < 4; ++i) {      // B tile: 64 rows x 8 float4
        const int idx = tid + i * 128;
        const int r = idx >> 3, q = idx & 7;
        float4 v = *(const float4*)(Q + (size_t)(colBlk + r) * N_SP + kg + q * 4);
        store_split4(sBhi, sBlo, r * LDS_STRIDE + q * 4, v);
      }
    } else {                             // K tail: q<4 valid, rest zero
#pragma unroll
      for (int i = 0; i < 8; ++i) {
        const int idx = tid + i * 128;
        const int r = idx >> 3, q = idx & 7;
        float4 v = (q < 4) ? *(const float4*)(Q + (size_t)(rowBlk + r) * N_SP + kg + q * 4)
                           : make_float4(0.f, 0.f, 0.f, 0.f);
        store_split4(sAhi, sAlo, r * LDS_STRIDE + q * 4, v);
      }
#pragma unroll
      for (int i = 0; i < 4; ++i) {
        const int idx = tid + i * 128;
        const int r = idx >> 3, q = idx & 7;
        float4 v = (q < 4) ? *(const float4*)(Q + (size_t)(colBlk + r) * N_SP + kg + q * 4)
                           : make_float4(0.f, 0.f, 0.f, 0.f);
        store_split4(sBhi, sBlo, r * LDS_STRIDE + q * 4, v);
      }
    }
    __syncthreads();

    v16bf ah[2], al[2], bh[4], bl[4];
#pragma unroll
    for (int m = 0; m < 2; ++m) {
      ah[m] = load_fragA(sAhi, wr + 16 * m, lane);
      al[m] = load_fragA(sAlo, wr + 16 * m, lane);
    }
#pragma unroll
    for (int n = 0; n < 4; ++n) {
      bh[n] = load_fragB(sBhi, 16 * n, lane);
      bl[n] = load_fragB(sBlo, 16 * n, lane);
    }
#pragma unroll
    for (int mi = 0; mi < 2; ++mi)
#pragma unroll
      for (int ni = 0; ni < 4; ++ni) {
        acc[mi][ni] = __builtin_amdgcn_wmma_f32_16x16x32_bf16(false, ah[mi], false, bh[ni], (short)0, acc[mi][ni], false, false);
        acc[mi][ni] = __builtin_amdgcn_wmma_f32_16x16x32_bf16(false, ah[mi], false, bl[ni], (short)0, acc[mi][ni], false, false);
        acc[mi][ni] = __builtin_amdgcn_wmma_f32_16x16x32_bf16(false, al[mi], false, bh[ni], (short)0, acc[mi][ni], false, false);
      }
    __syncthreads();
  }

  // C/D layout: VGPR i -> row (i + 8*(lane>=16)), col = lane&15
#pragma unroll
  for (int mi = 0; mi < 2; ++mi)
#pragma unroll
    for (int ni = 0; ni < 4; ++ni)
#pragma unroll
      for (int i = 0; i < 8; ++i) {
        const int r  = rowBlk + wr + 16 * mi + ((lane >> 4) << 3) + i;
        const int cc = colBlk + 16 * ni + (lane & 15);
        E[(size_t)r * C_CH + cc] = acc[mi][ni][i];
      }
}

// ---------------------------------------------------------------------------
// Kernel 2: row softmax of (rowmax - e) == exp(rowmin(e) - e)/sum, in place.
// ---------------------------------------------------------------------------
__global__ __launch_bounds__(256) void softmax_kernel(float* __restrict__ e) {
  __shared__ float red[256];
  float* p = e + (size_t)blockIdx.x * C_CH;
  const int tid = threadIdx.x;

  float v[4];
#pragma unroll
  for (int i = 0; i < 4; ++i) v[i] = p[tid + 256 * i];

  float mn = fminf(fminf(v[0], v[1]), fminf(v[2], v[3]));
  red[tid] = mn;
  __syncthreads();
  for (int off = 128; off > 0; off >>= 1) {
    if (tid < off) red[tid] = fminf(red[tid], red[tid + off]);
    __syncthreads();
  }
  mn = red[0];
  __syncthreads();

  float ex[4];
  float s = 0.0f;
#pragma unroll
  for (int i = 0; i < 4; ++i) { ex[i] = __expf(mn - v[i]); s += ex[i]; }
  red[tid] = s;
  __syncthreads();
  for (int off = 128; off > 0; off >>= 1) {
    if (tid < off) red[tid] += red[tid + off];
    __syncthreads();
  }
  const float inv = 1.0f / red[0];
#pragma unroll
  for (int i = 0; i < 4; ++i) p[tid + 256 * i] = ex[i] * inv;
}

// ---------------------------------------------------------------------------
// Kernel 3: out[b] = alpha * (attn[b] @ Q[b]) + x[b]
// grid (ceil(N/BN)=13, C/BM=8, B). B tile stored transposed into LDS.
// Column blocks 0..11 are interior (uniform unguarded path).
// ---------------------------------------------------------------------------
__global__ __launch_bounds__(128) void av_kernel(const float* __restrict__ attn,
                                                 const float* __restrict__ x,
                                                 const float* __restrict__ alpha,
                                                 float* __restrict__ out) {
  __shared__ __bf16 sAhi[BM * LDS_STRIDE];
  __shared__ __bf16 sAlo[BM * LDS_STRIDE];
  __shared__ __bf16 sBhi[BN * LDS_STRIDE];
  __shared__ __bf16 sBlo[BN * LDS_STRIDE];

  const int b = blockIdx.z;
  const float* A = attn + (size_t)b * C_CH * C_CH;
  const float* Q = x + (size_t)b * C_CH * N_SP;
  float* O = out + (size_t)b * C_CH * N_SP;
  const float alpha0 = alpha[0];

  const int rowBlk = blockIdx.y * BM;
  const int colBlk = blockIdx.x * BN;
  const bool fullBlk = (colBlk + BN) <= N_SP;   // uniform per block
  const int tid  = threadIdx.x;
  const int lane = tid & 31;
  const int wr   = (tid >> 5) * 32;

  v8f acc[2][4] = {};

  for (int kt = 0; kt < C_CH / 32; ++kt) {
    const int kg = kt * 32;
#pragma unroll
    for (int i = 0; i < 8; ++i) {        // A tile: 128 rows x 8 float4 (in bounds)
      const int idx = tid + i * 128;
      const int r = idx >> 3, q = idx & 7;
      float4 v = *(const float4*)(A + (size_t)(rowBlk + r) * C_CH + kg + q * 4);
      store_split4(sAhi, sAlo, r * LDS_STRIDE + q * 4, v);
    }
    // B tile: [32 k][64 n] from Q, stored transposed -> LDS [n][k]
    if (fullBlk) {
#pragma unroll
      for (int i = 0; i < 4; ++i) {
        const int idx = tid + i * 128;
        const int k = idx >> 4, nq = idx & 15;
        float4 v = *(const float4*)(Q + (size_t)(kg + k) * N_SP + colBlk + nq * 4);
        __bf16 h, l;
#pragma unroll
        for (int j = 0; j < 4; ++j) {
          const float f = (j == 0) ? v.x : (j == 1) ? v.y : (j == 2) ? v.z : v.w;
          h = (__bf16)f; l = (__bf16)(f - (float)h);
          sBhi[(nq * 4 + j) * LDS_STRIDE + k] = h;
          sBlo[(nq * 4 + j) * LDS_STRIDE + k] = l;
        }
      }
    } else {                              // last column block: 16 valid cols (nq<4)
#pragma unroll
      for (int i = 0; i < 4; ++i) {
        const int idx = tid + i * 128;
        const int k = idx >> 4, nq = idx & 15;
        float4 v = (nq < 4) ? *(const float4*)(Q + (size_t)(kg + k) * N_SP + colBlk + nq * 4)
                            : make_float4(0.f, 0.f, 0.f, 0.f);
        __bf16 h, l;
#pragma unroll
        for (int j = 0; j < 4; ++j) {
          const float f = (j == 0) ? v.x : (j == 1) ? v.y : (j == 2) ? v.z : v.w;
          h = (__bf16)f; l = (__bf16)(f - (float)h);
          sBhi[(nq * 4 + j) * LDS_STRIDE + k] = h;
          sBlo[(nq * 4 + j) * LDS_STRIDE + k] = l;
        }
      }
    }
    __syncthreads();

    v16bf ah[2], al[2], bh[4], bl[4];
#pragma unroll
    for (int m = 0; m < 2; ++m) {
      ah[m] = load_fragA(sAhi, wr + 16 * m, lane);
      al[m] = load_fragA(sAlo, wr + 16 * m, lane);
    }
#pragma unroll
    for (int n = 0; n < 4; ++n) {
      bh[n] = load_fragB(sBhi, 16 * n, lane);
      bl[n] = load_fragB(sBlo, 16 * n, lane);
    }
#pragma unroll
    for (int mi = 0; mi < 2; ++mi)
#pragma unroll
      for (int ni = 0; ni < 4; ++ni) {
        acc[mi][ni] = __builtin_amdgcn_wmma_f32_16x16x32_bf16(false, ah[mi], false, bh[ni], (short)0, acc[mi][ni], false, false);
        acc[mi][ni] = __builtin_amdgcn_wmma_f32_16x16x32_bf16(false, ah[mi], false, bl[ni], (short)0, acc[mi][ni], false, false);
        acc[mi][ni] = __builtin_amdgcn_wmma_f32_16x16x32_bf16(false, al[mi], false, bh[ni], (short)0, acc[mi][ni], false, false);
      }
    __syncthreads();
  }

#pragma unroll
  for (int mi = 0; mi < 2; ++mi)
#pragma unroll
    for (int ni = 0; ni < 4; ++ni)
#pragma unroll
      for (int i = 0; i < 8; ++i) {
        const int r  = rowBlk + wr + 16 * mi + ((lane >> 4) << 3) + i;
        const int cc = colBlk + 16 * ni + (lane & 15);
        if (cc < N_SP) {
          const size_t off = (size_t)r * N_SP + cc;
          O[off] = alpha0 * acc[mi][ni][i] + Q[off];
        }
      }
}

// ---------------------------------------------------------------------------
extern "C" void kernel_launch(void* const* d_in, const int* in_sizes, int n_in,
                              void* d_out, int out_size, void* d_ws, size_t ws_size,
                              hipStream_t stream) {
  const float* x     = (const float*)d_in[0];
  const float* alpha = (const float*)d_in[1];
  float* out         = (float*)d_out;
  float* energy      = (float*)d_ws;     // B*C*C*4 = 128 MB scratch

  dim3 g1(C_CH / BN, C_CH / BM, B_BATCH);            // 16 x 8 x 32
  qqt_kernel<<<g1, 128, 0, stream>>>(x, energy);

  softmax_kernel<<<dim3(B_BATCH * C_CH), 256, 0, stream>>>(energy);

  dim3 g3((N_SP + BN - 1) / BN, C_CH / BM, B_BATCH); // 13 x 8 x 32
  av_kernel<<<g3, 128, 0, stream>>>(energy, x, alpha, out);
}
